// Head_58368605553016
// MI455X (gfx1250) — matrix operand order repaired and checked by reading, not compiled, not run on hardware
//
#include <hip/hip_runtime.h>
#include <hip/hip_bf16.h>
#include <cstdint>

#define USE_TDM 1

typedef __attribute__((ext_vector_type(16))) _Float16     v16h;
typedef __attribute__((ext_vector_type(8)))  float        v8f;
typedef __attribute__((ext_vector_type(4)))  unsigned int u32x4;
typedef __attribute__((ext_vector_type(8)))  int          i32x8;
typedef __attribute__((ext_vector_type(4)))  int          i32x4;

#define HEAD 64
#define TSEQ 4096
#define CDIM 1024
#define QLD  72   // Qs row stride (halves): 64 + 4-DWORD TDM pad
#define VLD  68   // VT row stride (halves)
#define PLD  72   // Pb row stride (halves)

// ===================== Kernel 1: projections K,Q,V = x @ W ===================
__global__ void __launch_bounds__(256)
proj_kernel(const float* __restrict__ x,
            const float* __restrict__ Wk,
            const float* __restrict__ Wq,
            const float* __restrict__ Wv,
            _Float16* __restrict__ Kp,
            _Float16* __restrict__ Qp,
            _Float16* __restrict__ Vp) {
    __shared__ _Float16 WT[HEAD * 36];   // W tile transposed [h][kk]

    const int tid   = threadIdx.x;
    const int lane  = tid & 31;
    const int wave  = tid >> 5;
    const int half8 = (lane < 16) ? 0 : 8;
    const int koffA = (lane < 16) ? 0 : 8;
    const int koffB = (lane < 16) ? 0 : 16;

    const float* W = (blockIdx.y == 0) ? Wk : (blockIdx.y == 1) ? Wq : Wv;
    _Float16*    P = (blockIdx.y == 0) ? Kp : (blockIdx.y == 1) ? Qp : Vp;

    const int row0 = blockIdx.x * 128 + wave * 16;

    const v8f vzero = {0.f,0.f,0.f,0.f,0.f,0.f,0.f,0.f};
    v8f acc[4];
    #pragma unroll
    for (int j = 0; j < 4; ++j) acc[j] = vzero;

    for (int k0 = 0; k0 < CDIM; k0 += 32) {
        __syncthreads();
        // cooperative: W[k0..k0+31][0..63] f32 -> WT[h][kk] f16 (b128 loads)
        for (int c = tid; c < 512; c += 256) {
            int kk = c >> 4, h0 = (c & 15) * 4;
            const float4 w = *(const float4*)&W[(size_t)(k0 + kk) * HEAD + h0];
            WT[(h0 + 0) * 36 + kk] = (_Float16)w.x;
            WT[(h0 + 1) * 36 + kk] = (_Float16)w.y;
            WT[(h0 + 2) * 36 + kk] = (_Float16)w.z;
            WT[(h0 + 3) * 36 + kk] = (_Float16)w.w;
        }
        __syncthreads();

        const float* xr = x + (size_t)(row0 + (lane & 15)) * CDIM + k0;
        v16h a;
        #pragma unroll
        for (int i = 0; i < 8; ++i) {
            a[i]     = (_Float16)xr[koffA + i];
            a[8 + i] = (_Float16)xr[16 + koffA + i];
        }
        #pragma unroll
        for (int j = 0; j < 4; ++j) {
            const _Float16* wr = &WT[((lane & 15) + 16 * j) * 36];
            v16h b;
            #pragma unroll
            for (int i = 0; i < 8; ++i) { b[i] = wr[koffA + i]; b[8 + i] = wr[16 + koffA + i]; }
            acc[j] = __builtin_amdgcn_wmma_f32_16x16x32_f16(
                         false, a, false, b, (short)0, acc[j], false, false);
        }
    }
    #pragma unroll
    for (int j = 0; j < 4; ++j) {
        int h = (lane & 15) + 16 * j;
        #pragma unroll
        for (int r = 0; r < 8; ++r)
            P[(size_t)(row0 + r + half8) * HEAD + h] = (_Float16)acc[j][r];
    }
}

#if USE_TDM
// TDM: load a dense 64x64 f16 tile (row stride 64) into LDS with padding
// (4 DWORDs inserted after every 32 DWORDs -> LDS row stride = 72 halves).
__device__ static inline void tdm_load_tile64(const _Float16* gsrc, unsigned lds_off) {
    unsigned long long ga = (unsigned long long)(uintptr_t)gsrc;
    u32x4 g0 = { 1u,                                   // count=1
                 lds_off,                              // lds_addr
                 (unsigned)(ga & 0xFFFFFFFFu),         // global_addr lo
                 ((unsigned)((ga >> 32) & 0x1FFFFFFu)) | (2u << 30) }; // hi | type=2
    i32x8 g1 = { (int)((1u << 16)            // data_size = 1 (2 bytes)
                      | (1u << 20)           // pad_enable
                      | (4u << 22)           // pad_interval: every 32 DWORDs
                      | (3u << 25)),         // pad_amount: 4 DWORDs
                 (int)(64u << 16),           // tensor_dim0 = 64
                 (int)(64u << 16),           // tensor_dim1 = 64
                 (int)(64u << 16),           // tile_dim0 = 64
                 64,                         // tile_dim1 = 64
                 64,                         // tensor_dim0_stride = 64
                 0, 0 };
    i32x4 gz = { 0, 0, 0, 0 };
#if __has_include(<hip/amd_detail/amd_gfx1250_TDM.h>)
    i32x8 gz8 = { 0, 0, 0, 0, 0, 0, 0, 0 };
    __builtin_amdgcn_tensor_load_to_lds(g0, g1, gz, gz, gz8, 0);
#else
    __builtin_amdgcn_tensor_load_to_lds(g0, g1, gz, gz, 0);
#endif
}
#endif

// ===================== Kernel 2: fused causal flash attention ================
// wei[t,s] = k[t].q[s] * H^-0.5 ; softmax over s<=t ; @ v.  64-wide s-blocks.
__global__ void __launch_bounds__(256)
attn_kernel(const _Float16* __restrict__ Kp,
            const _Float16* __restrict__ Qp,
            const _Float16* __restrict__ Vp,
            float* __restrict__ out) {
    __shared__ __align__(16) _Float16 Qs[64 * QLD];    // q s-block [kk][h]
    __shared__ _Float16 VT[HEAD * VLD];                // v s-block transposed [h][kk]
    __shared__ __align__(16) _Float16 Pb[8 * 16 * PLD];// per-wave P tiles

    const int tid   = threadIdx.x;
    const int lane  = tid & 31;
    const int wave  = tid >> 5;
    const int half8 = (lane < 16) ? 0 : 8;
    const int koffA = (lane < 16) ? 0 : 8;
    const int koffB = (lane < 16) ? 0 : 16;

    const int batch = blockIdx.x >> 5;
    const int trel0 = (blockIdx.x & 31) * 128 + wave * 16;
    const size_t base = (size_t)batch * TSEQ * HEAD;
    const _Float16* Kb = Kp + base;
    const _Float16* Qb = Qp + base;
    const _Float16* Vb = Vp + base;

    // A operand: this wave's 16 rows of the k-projection (K-dim 64 = 2 chunks)
    v16h aK[2];
    {
        const _Float16* kr = Kb + (size_t)(trel0 + (lane & 15)) * HEAD;
        #pragma unroll
        for (int c = 0; c < 2; ++c)
            #pragma unroll
            for (int i = 0; i < 8; ++i) {
                aK[c][i]     = kr[c * 32 + koffA + i];
                aK[c][8 + i] = kr[c * 32 + 16 + koffA + i];
            }
    }

    const v8f vzero = {0.f,0.f,0.f,0.f,0.f,0.f,0.f,0.f};
    v8f O[4];
    #pragma unroll
    for (int j = 0; j < 4; ++j) O[j] = vzero;
    float mrun[8], lrun[8];
    #pragma unroll
    for (int r = 0; r < 8; ++r) { mrun[r] = -3.0e38f; lrun[r] = 0.0f; }

    const int tmax_block = (blockIdx.x & 31) * 128 + 127;
#if USE_TDM
    const unsigned qs_off = (unsigned)(uintptr_t)(void*)Qs;
#endif

    for (int s0 = 0; s0 <= tmax_block; s0 += 64) {
        __syncthreads();   // prior iteration's LDS reads complete
#if USE_TDM
        if (wave == 0) tdm_load_tile64(Qb + (size_t)s0 * HEAD, qs_off);
#else
        {   // manual b128 staging of Q (row stride QLD in LDS)
            const uint4* src = (const uint4*)(Qb + (size_t)s0 * HEAD);
            for (int c = tid; c < 512; c += 256) {
                int kk = c >> 3, h0 = (c & 7) * 8;
                *(uint4*)&Qs[kk * QLD + h0] = src[c];
            }
        }
#endif
        // V staging: b128 global loads, transposed scatter into LDS
        for (int c = tid; c < 512; c += 256) {
            int kk = c >> 3, h0 = (c & 7) * 8;
            uint4 d = *(const uint4*)(Vb + (size_t)(s0 + kk) * HEAD + h0);
            const _Float16* t = (const _Float16*)&d;
            #pragma unroll
            for (int e = 0; e < 8; ++e) VT[(h0 + e) * VLD + kk] = t[e];
        }
        // prefetch next s-block while this one is consumed
        const int s1 = s0 + 64;
        if (s1 <= tmax_block) {
            if (tid < 32)      __builtin_prefetch(Qb + (size_t)s1 * HEAD + tid * 128, 0, 1);
            else if (tid < 64) __builtin_prefetch(Vb + (size_t)s1 * HEAD + (size_t)(tid - 32) * 128, 0, 1);
        }
#if USE_TDM
        if (wave == 0) __builtin_amdgcn_s_wait_tensorcnt((short)0);
#endif
        __syncthreads();

        const bool active = (s0 <= trel0 + 15);   // wave-uniform
        if (active) {
            // S = K_rows @ Q^T : four 16-col n-tiles, K-dim 64 = 2 chunks
            v8f S[4];
            #pragma unroll
            for (int nt = 0; nt < 4; ++nt) {
                v8f acc = vzero;
                #pragma unroll
                for (int c = 0; c < 2; ++c) {
                    const _Float16* qr = &Qs[(nt * 16 + (lane & 15)) * QLD + c * 32];
                    v16h b;
                    #pragma unroll
                    for (int i = 0; i < 8; ++i) { b[i] = qr[koffA + i]; b[8 + i] = qr[16 + koffA + i]; }
                    acc = __builtin_amdgcn_wmma_f32_16x16x32_f16(
                              false, aK[c], false, b, (short)0, acc, false, false);
                }
                S[nt] = acc;
            }
            // online softmax over 64-wide s-block (row m lives across 16 lanes)
            _Float16* Pw = &Pb[wave * 16 * PLD];
            #pragma unroll
            for (int r = 0; r < 8; ++r) {
                const int trow = trel0 + r + half8;
                const int sc   = s0 + (lane & 15);
                float v[4];
                #pragma unroll
                for (int nt = 0; nt < 4; ++nt) {
                    v[nt] = S[nt][r] * 0.125f;               // HEAD^-0.5
                    if (sc + 16 * nt > trow) v[nt] = -1.0e30f;
                }
                float mx = fmaxf(fmaxf(v[0], v[1]), fmaxf(v[2], v[3]));
                #pragma unroll
                for (int d = 1; d < 16; d <<= 1) mx = fmaxf(mx, __shfl_xor(mx, d, 32));
                float mnew  = fmaxf(mrun[r], mx);
                float alpha = __expf(mrun[r] - mnew);
                float p[4], sum = 0.0f;
                #pragma unroll
                for (int nt = 0; nt < 4; ++nt) { p[nt] = __expf(v[nt] - mnew); sum += p[nt]; }
                #pragma unroll
                for (int d = 1; d < 16; d <<= 1) sum += __shfl_xor(sum, d, 32);
                lrun[r] = lrun[r] * alpha + sum;
                mrun[r] = mnew;
                #pragma unroll
                for (int j = 0; j < 4; ++j) O[j][r] *= alpha;
                #pragma unroll
                for (int nt = 0; nt < 4; ++nt)
                    Pw[(r + half8) * PLD + 16 * nt + (lane & 15)] = (_Float16)p[nt];
            }
        }
        __syncthreads();   // P visible; uniform barrier count across waves
        if (active) {
            // O += P(16x64) @ V(64x64)
            const _Float16* pr = &Pb[wave * 16 * PLD + (lane & 15) * PLD];
            v16h aP[2];
            #pragma unroll
            for (int c = 0; c < 2; ++c)
                #pragma unroll
                for (int i = 0; i < 8; ++i) {
                    aP[c][i]     = pr[c * 32 + koffA + i];
                    aP[c][8 + i] = pr[c * 32 + 16 + koffA + i];
                }
            #pragma unroll
            for (int j = 0; j < 4; ++j) {
                #pragma unroll
                for (int c = 0; c < 2; ++c) {
                    const _Float16* vr = &VT[((lane & 15) + 16 * j) * VLD + c * 32];
                    v16h b;
                    #pragma unroll
                    for (int i = 0; i < 8; ++i) { b[i] = vr[koffA + i]; b[8 + i] = vr[16 + koffA + i]; }
                    O[j] = __builtin_amdgcn_wmma_f32_16x16x32_f16(
                               false, aP[c], false, b, (short)0, O[j], false, false);
                }
            }
        }
    }

    // normalize by softmax denominator and store f32 output
    #pragma unroll
    for (int j = 0; j < 4; ++j) {
        int h = (lane & 15) + 16 * j;
        #pragma unroll
        for (int r = 0; r < 8; ++r) {
            out[base + (size_t)(trel0 + r + half8) * HEAD + h] = O[j][r] / lrun[r];
        }
    }
}

extern "C" void kernel_launch(void* const* d_in, const int* in_sizes, int n_in,
                              void* d_out, int out_size, void* d_ws, size_t ws_size,
                              hipStream_t stream) {
    const float* x  = (const float*)d_in[0];   // [4,4096,1024]
    const float* Wk = (const float*)d_in[1];   // [1024,64]
    const float* Wq = (const float*)d_in[2];
    const float* Wv = (const float*)d_in[3];
    float* out = (float*)d_out;                // [4,4096,64]

    const size_t rows = (size_t)4 * TSEQ;      // 16384
    _Float16* Kp = (_Float16*)d_ws;            // 2 MB each in workspace
    _Float16* Qp = Kp + rows * HEAD;
    _Float16* Vp = Qp + rows * HEAD;

    dim3 pg((unsigned)(rows / 128), 3);
    proj_kernel<<<pg, 256, 0, stream>>>(x, Wk, Wq, Wv, Kp, Qp, Vp);

    dim3 ag((unsigned)(rows / 128));
    attn_kernel<<<ag, 256, 0, stream>>>(Kp, Qp, Vp, out);
}
